// Cabasc_74474732912737
// MI455X (gfx1250) — matrix operand order repaired and checked by reading, not compiled, not run on hardware
//
#include <hip/hip_runtime.h>
#include <cmath>

typedef __attribute__((ext_vector_type(16))) _Float16 v16h;
typedef __attribute__((ext_vector_type(8)))  _Float16 v8h;
typedef __attribute__((ext_vector_type(4)))  _Float16 v4h;
typedef __attribute__((ext_vector_type(8)))  float    v8f;

constexpr int kV = 50000, kE = 300, kP = 3, kB = 128, kS = 512, kA = 8;
constexpr int NT     = 19;        // N tiles of 16 covering 304 >= 300
constexpr int KSTEPS = 10;        // K steps of 32 covering 320 >= 300
constexpr int KPAD   = 320;
constexpr int NPAD   = 304;

// workspace layout (bytes, 256-aligned)
constexpr size_t OFF_MEMLEN = 0;                                  // 128 ints
constexpr size_t OFF_VS     = 1024;                               // 128*300 f32
constexpr size_t OFF_C      = OFF_VS + (size_t)kB * kE * 4;       // 128*304 f32 (zero tail)
constexpr size_t OFF_SCORES = OFF_C + (size_t)kB * NPAD * 4;      // 128*512 f32 (-> alpha)
constexpr size_t OFF_W2PAD  = OFF_SCORES + (size_t)kB * kS * 4;   // 304 f32 zero-padded
constexpr size_t OFF_W1H    = OFF_W2PAD + 2048;                   // packed fp16 W1m frags

// Branch-free tanh: tanh(x) = 1 - 2/(exp(2x)+1).
// v_exp + v_rcp + 3 VALU, saturates correctly at +/-inf, no exec divergence.
__device__ __forceinline__ float fast_tanh(float x) {
  const float e = __expf(2.0f * x);
  const float r = __builtin_amdgcn_rcpf(e + 1.0f);
  return __builtin_fmaf(-2.0f, r, 1.0f);
}

// ---------------------------------------------------------------------------
// Kernel 1: per-batch lengths, v_a, v_s, and c = v_a@W1a + v_s@W1s + b1
// ---------------------------------------------------------------------------
__global__ __launch_bounds__(320) void k_prep(const int* __restrict__ text,
                                              const int* __restrict__ aspect,
                                              const float* __restrict__ embed,
                                              const float* __restrict__ W1,
                                              const float* __restrict__ b1,
                                              int* __restrict__ memLen,
                                              float* __restrict__ vsOut,
                                              float* __restrict__ cOut) {
  const int b = blockIdx.x, t = threadIdx.x;
  __shared__ int sIdx[kS];
  __shared__ int aIdx[kA];
  __shared__ int cnt, acnt;
  __shared__ float vaS[kE], vsS[kE];
  if (t == 0) { cnt = 0; acnt = 0; }
  for (int s = t; s < kS; s += 320) sIdx[s] = text[b * kS + s];
  if (t < kA) aIdx[t] = aspect[b * kA + t];
  __syncthreads();
  int lc = 0;
  for (int s = t; s < kS; s += 320) lc += (sIdx[s] != 0);
  if (lc) atomicAdd(&cnt, lc);
  if (t < kA && aIdx[t] != 0) atomicAdd(&acnt, 1);
  __syncthreads();
  const int ml = cnt;
  const float invml = 1.0f / (float)ml;
  const float inval = 1.0f / (float)acnt;
  if (t == 0) memLen[b] = ml;
  if (t < kE) {
    float accA = 0.f;
#pragma unroll
    for (int a = 0; a < kA; ++a) accA += embed[(size_t)aIdx[a] * kE + t];
    float accS = 0.f;
    for (int s = 0; s < ml; ++s) accS += embed[(size_t)sIdx[s] * kE + t];
    const float va = accA * inval;
    const float vs = accS * invml;
    vaS[t] = va; vsS[t] = vs;
    vsOut[b * kE + t] = vs;
  }
  __syncthreads();
  if (t < NPAD) {
    float c = 0.f;
    if (t < kE) {
      c = b1[t];
      for (int k = 0; k < kE; ++k)
        c += vaS[k] * W1[(size_t)(kE + k) * kE + t] +
             vsS[k] * W1[(size_t)(2 * kE + k) * kE + t];
    }
    cOut[b * NPAD + t] = c;   // zero for padded columns 300..303
  }
}

// ---------------------------------------------------------------------------
// Kernel 2: pack W1m into fp16 WMMA-B fragments (zero-padded) + padded w2.
// Fragment f = ks*NT + ntile: 32 lanes x 16 halfs; half i -> K = ks*32 +
// (lane/16)*16 + i; column n = ntile*16 + lane%16.  Lane's v16h = contiguous
// 32 bytes -> coalesced b128 pair loads in k_scores.
// ---------------------------------------------------------------------------
__global__ __launch_bounds__(256) void k_packW(const float* __restrict__ W1,
                                               const float* __restrict__ w2,
                                               _Float16* __restrict__ W1h,
                                               float* __restrict__ w2pad) {
  const int gid = blockIdx.x * 256 + threadIdx.x;
  if (gid < NPAD) w2pad[gid] = (gid < kE) ? w2[gid] : 0.f;
  const int total = KSTEPS * NT * 32 * 16;
  if (gid >= total) return;
  const int i    = gid & 15;
  const int lane = (gid >> 4) & 31;
  const int f    = gid >> 9;
  const int ks = f / NT, ntile = f % NT;
  const int n = ntile * 16 + (lane & 15);
  const int k = ks * 32 + (lane >> 4) * 16 + i;
  const float v = (k < kE && n < kE) ? W1[(size_t)k * kE + n] : 0.f;
  W1h[gid] = (_Float16)v;
}

// ---------------------------------------------------------------------------
// Kernel 3: fused GEMM + tanh + dot(w2) -> scores.  One block = 128 rows of
// "memory" staged to LDS as fp16 (decay applied); 8 waves x one 16-row tile.
// ---------------------------------------------------------------------------
__global__ __launch_bounds__(256) void k_scores(const int* __restrict__ text,
                                                const float* __restrict__ embed,
                                                const int* __restrict__ memLen,
                                                const float* __restrict__ cbuf,
                                                const float* __restrict__ w2pad,
                                                const _Float16* __restrict__ W1h,
                                                float* __restrict__ scores) {
  const int tid  = threadIdx.x;
  const int lane = tid & 31;
  const int wave = tid >> 5;
  const int rowBase = blockIdx.x * 128;       // flat row = b*512 + s
  const int b = rowBase / kS;
  const int sBase = rowBase % kS;

  __shared__ int sIdx[128];
  __shared__ _Float16 As[128 * KPAD];         // 80 KB
  if (tid < 128) sIdx[tid] = text[rowBase + tid];
  __syncthreads();
  const int ml = memLen[b];
  const float invml = 1.0f / (float)ml;

  // Vectorized staging: float4 per step, 40 iterations/thread.
  constexpr int QK = KPAD / 4;                // 80 quads per row
  for (int q = tid; q < 128 * QK; q += 256) {
    const int row = q / QK;
    const int kq  = (q - row * QK) * 4;
    const int s = sBase + row;
    float4 v = make_float4(0.f, 0.f, 0.f, 0.f);
    if (kq < kE && s < ml) {                  // kE % 4 == 0: quad fully valid
      const float decay = 1.f - (float)s * invml;
      const float4 src = *(const float4*)(embed + (size_t)sIdx[row] * kE + kq);
      v.x = src.x * decay; v.y = src.y * decay;
      v.z = src.z * decay; v.w = src.w * decay;
    }
    v4h hv = { (_Float16)v.x, (_Float16)v.y, (_Float16)v.z, (_Float16)v.w };
    *(v4h*)(&As[(size_t)row * KPAD + kq]) = hv;
  }
  __syncthreads();

  const v16h* __restrict__ Bp = (const v16h*)W1h;
  const int m = lane & 15;    // A row within tile / C column index
  const int g = lane >> 4;    // half-wave group

  float scoreAcc[8];
#pragma unroll
  for (int r = 0; r < 8; ++r) scoreAcc[r] = 0.f;

  for (int ntile = 0; ntile < NT; ++ntile) {
    v8f acc = {};
#pragma unroll 2
    for (int ks = 0; ks < KSTEPS; ++ks) {
      // A fragment (contiguous K order, matching packed B): g=0 holds K {0..15},
      // g=1 holds K {16..31} of this K-step.
      const _Float16* ap = &As[(size_t)(wave * 16 + m) * KPAD + ks * 32 + g * 16];
      const v8h lo = *(const v8h*)ap;
      const v8h hi = *(const v8h*)(ap + 8);
      const v16h a = __builtin_shufflevector(lo, hi,
          0, 1, 2, 3, 4, 5, 6, 7, 8, 9, 10, 11, 12, 13, 14, 15);
      const v16h bfrag = Bp[(size_t)(ks * NT + ntile) * 32 + lane];
      acc = __builtin_amdgcn_wmma_f32_16x16x32_f16(false, a, false, bfrag,
                                                   (short)0, acc, false, false);
    }
    // Epilogue: lane-local accumulation only; cross-lane reduce ONCE at end.
    const int n = ntile * 16 + m;
    const float cn  = cbuf[b * NPAD + n];
    const float w2n = w2pad[n];
#pragma unroll
    for (int r = 0; r < 8; ++r)
      scoreAcc[r] += fast_tanh(acc[r] + cn) * w2n;
  }

  // Reduce over the 16 lanes holding the N columns (per half-wave group).
#pragma unroll
  for (int r = 0; r < 8; ++r) {
    float v = scoreAcc[r];
    v += __shfl_xor(v, 1, 32);
    v += __shfl_xor(v, 2, 32);
    v += __shfl_xor(v, 4, 32);
    v += __shfl_xor(v, 8, 32);
    scoreAcc[r] = v;
  }
  if (m == 0) {
#pragma unroll
    for (int r = 0; r < 8; ++r)
      scores[rowBase + wave * 16 + g * 8 + r] = scoreAcc[r];
  }
}

// ---------------------------------------------------------------------------
// Kernel 4: softmax over S=512 per batch (in place: scores -> alpha)
// ---------------------------------------------------------------------------
__global__ __launch_bounds__(256) void k_softmax(float* __restrict__ scores) {
  const int b = blockIdx.x, t = threadIdx.x;
  __shared__ float red[256];
  const float a0 = scores[b * kS + t];
  const float a1 = scores[b * kS + t + 256];
  red[t] = fmaxf(a0, a1);
  __syncthreads();
  for (int off = 128; off; off >>= 1) {
    if (t < off) red[t] = fmaxf(red[t], red[t + off]);
    __syncthreads();
  }
  const float mx = red[0];
  __syncthreads();
  const float e0 = __expf(a0 - mx), e1 = __expf(a1 - mx);
  red[t] = e0 + e1;
  __syncthreads();
  for (int off = 128; off; off >>= 1) {
    if (t < off) red[t] += red[t + off];
    __syncthreads();
  }
  const float inv = 1.f / red[0];
  scores[b * kS + t] = e0 * inv;
  scores[b * kS + t + 256] = e1 * inv;
}

// ---------------------------------------------------------------------------
// Kernel 5: v_ts (alpha-weighted gather), v_ns, v_ms = tanh(@Wm+bm), out=@Wd+bd
// ---------------------------------------------------------------------------
__global__ __launch_bounds__(320) void k_final(const int* __restrict__ text,
                                               const float* __restrict__ embed,
                                               const int* __restrict__ memLen,
                                               const float* __restrict__ vsBuf,
                                               const float* __restrict__ alpha,
                                               const float* __restrict__ Wm,
                                               const float* __restrict__ bm,
                                               const float* __restrict__ Wd,
                                               const float* __restrict__ bd,
                                               float* __restrict__ out) {
  const int b = blockIdx.x, t = threadIdx.x;
  __shared__ int sIdx[kS];
  __shared__ float al[kS];
  __shared__ float vns[kE];
  __shared__ float vms[kE];
  for (int s = t; s < kS; s += 320) {
    sIdx[s] = text[b * kS + s];
    al[s]   = alpha[b * kS + s];
  }
  __syncthreads();
  const int ml = memLen[b];
  const float invml = 1.f / (float)ml;
  if (t < kE) {
    float acc = 0.f;
    for (int s = 0; s < ml; ++s) {
      const float w = al[s] * (1.f - (float)s * invml);
      acc += embed[(size_t)sIdx[s] * kE + t] * w;
    }
    vns[t] = acc + vsBuf[b * kE + t];
  }
  __syncthreads();
  if (t < kE) {
    float h = bm[t];
    for (int k = 0; k < kE; ++k) h += vns[k] * Wm[(size_t)k * kE + t];
    vms[t] = fast_tanh(h);
  }
  __syncthreads();
  if (t < kP) {
    float o = bd[t];
    for (int e = 0; e < kE; ++e) o += vms[e] * Wd[(size_t)e * kP + t];
    out[b * kP + t] = o;
  }
}

// ---------------------------------------------------------------------------
extern "C" void kernel_launch(void* const* d_in, const int* in_sizes, int n_in,
                              void* d_out, int out_size, void* d_ws, size_t ws_size,
                              hipStream_t stream) {
  const int*   text   = (const int*)d_in[0];
  const int*   aspect = (const int*)d_in[1];
  // d_in[2]=x_l, d_in[3]=x_r are unused by the reference
  const float* embed  = (const float*)d_in[4];
  const float* W1     = (const float*)d_in[5];
  const float* b1     = (const float*)d_in[6];
  const float* w2     = (const float*)d_in[7];
  const float* Wm     = (const float*)d_in[8];
  const float* bm     = (const float*)d_in[9];
  const float* Wd     = (const float*)d_in[10];
  const float* bd     = (const float*)d_in[11];
  float* out = (float*)d_out;

  char* ws = (char*)d_ws;
  int*      memLen = (int*)     (ws + OFF_MEMLEN);
  float*    vsBuf  = (float*)   (ws + OFF_VS);
  float*    cBuf   = (float*)   (ws + OFF_C);
  float*    scores = (float*)   (ws + OFF_SCORES);
  float*    w2pad  = (float*)   (ws + OFF_W2PAD);
  _Float16* W1h    = (_Float16*)(ws + OFF_W1H);

  k_prep<<<kB, 320, 0, stream>>>(text, aspect, embed, W1, b1, memLen, vsBuf, cBuf);

  const int packTotal = KSTEPS * NT * 32 * 16;
  k_packW<<<(packTotal + 255) / 256, 256, 0, stream>>>(W1, w2, W1h, w2pad);

  k_scores<<<(kB * kS) / 128, 256, 0, stream>>>(text, embed, memLen, cBuf,
                                                w2pad, W1h, scores);

  k_softmax<<<kB, 256, 0, stream>>>(scores);

  k_final<<<kB, 320, 0, stream>>>(text, embed, memLen, vsBuf, scores,
                                  Wm, bm, Wd, bd, out);
}
